// GNN_24927990186434
// MI455X (gfx1250) — matrix operand order repaired and checked by reading, not compiled, not run on hardware
//
#include <hip/hip_runtime.h>

// CDNA5 / gfx1250 GNN (TAGConv) implementation.
// GEMMs run on v_wmma_f32_16x16x32_f16 (f16 in, f32 accumulate).
// Segment sums use global_atomic_add_f32; gathers are b128 row loads.

typedef __attribute__((ext_vector_type(16))) _Float16 v16h;
typedef __attribute__((ext_vector_type(8)))  float    v8f;

#define GNN_N_NODES  100000
#define GNN_NODE_DIM 32
#define GNN_EDGE_DIM 32
#define GNN_HIDDEN   64

__device__ __forceinline__ v8f wmma_f16(v16h a, v16h b, v8f c) {
  // 8 args: (neg_a, A, neg_b, B, c_mod, C, reuse_a, reuse_b)
  return __builtin_amdgcn_wmma_f32_16x16x32_f16(false, a, false, b, (short)0, c,
                                                false, false);
}

// Load one K=32 chunk of the A fragment for this lane from a contiguous
// 32-float row. A layout (16-bit A 16x32): lane holds row M=lane%16;
// lanes 0-15: halves j=0..7 -> K=j,        j=8..15 -> K=16+(j-8)
// lanes16-31: halves j=0..7 -> K=8+j,      j=8..15 -> K=24+(j-8)
__device__ __forceinline__ v16h load_afrag32(const float* __restrict__ rowp, bool hi) {
  const int o = hi ? 8 : 0;
  float4 f0 = *(const float4*)(rowp + o);
  float4 f1 = *(const float4*)(rowp + o + 4);
  float4 f2 = *(const float4*)(rowp + o + 16);
  float4 f3 = *(const float4*)(rowp + o + 20);
  v16h a;
  a[0]=(_Float16)f0.x;  a[1]=(_Float16)f0.y;  a[2]=(_Float16)f0.z;  a[3]=(_Float16)f0.w;
  a[4]=(_Float16)f1.x;  a[5]=(_Float16)f1.y;  a[6]=(_Float16)f1.z;  a[7]=(_Float16)f1.w;
  a[8]=(_Float16)f2.x;  a[9]=(_Float16)f2.y;  a[10]=(_Float16)f2.z; a[11]=(_Float16)f2.w;
  a[12]=(_Float16)f3.x; a[13]=(_Float16)f3.y; a[14]=(_Float16)f3.z; a[15]=(_Float16)f3.w;
  return a;
}

// Stage W (K x 64, row-major, fp32) into LDS as pre-swizzled f16 B fragments.
// B layout (16-bit B 32x16): lane holds col N=lane%16;
// lanes 0-15: j=0..7 -> K=j, j=8..15 -> K=16+(j-8); lanes 16-31: j=0..7 -> K=8+j, j=8..15 -> K=24+(j-8).
// Fragment (kc, nt) stored contiguously: 32 lanes x 16 halves (32 B per lane).
__device__ __forceinline__ void stage_bfrags(_Float16* lds, const float* __restrict__ W,
                                             int KC, int tid, int nthreads) {
  for (int p = tid; p < KC * 4 * 32; p += nthreads) {
    const int lane = p & 31;
    const int frag = p >> 5;
    const int kc = frag >> 2;
    const int nt = frag & 3;
    const int n  = nt * 16 + (lane & 15);
    const int kb = kc * 32 + ((lane & 16) ? 8 : 0);
    _Float16* o = lds + (size_t)p * 16;
#pragma unroll
    for (int j = 0; j < 16; ++j) o[j] = (_Float16)W[(kb + j + (j >= 8 ? 8 : 0)) * GNN_HIDDEN + n];
  }
}

// ---------------------------------------------------------------------------
// Edge message kernel: m = relu([fea[src], efea] @ Wmsg + bmsg), 64-wide,
// scatter-added into X0[:, 32:96] (h_neigh columns).
// One wave handles a tile of 16 edges; K = 64 (2 chunks of 32).
// ---------------------------------------------------------------------------
__global__ void __launch_bounds__(128)
msg_kernel(const float* __restrict__ fea, const float* __restrict__ efea,
           const int* __restrict__ src, const int* __restrict__ dst,
           const float* __restrict__ Wmsg, const float* __restrict__ bmsg,
           float* __restrict__ X0, int n_edges) {
  __shared__ _Float16 Wf[2 * 4 * 32 * 16];  // 8 KB
  stage_bfrags(Wf, Wmsg, 2, threadIdx.x, blockDim.x);
  __syncthreads();

  const int  lane = threadIdx.x & 31;
  const int  row  = lane & 15;
  const bool hi   = lane >= 16;
  const int  wave = blockIdx.x * (blockDim.x >> 5) + (threadIdx.x >> 5);
  const int  nwav = gridDim.x * (blockDim.x >> 5);
  const int  ntiles = n_edges >> 4;  // n_edges is a multiple of 16

  // B fragments live in registers for the whole kernel (8 x v16h).
  v16h b[2][4];
#pragma unroll
  for (int kc = 0; kc < 2; ++kc)
#pragma unroll
    for (int nt = 0; nt < 4; ++nt)
      b[kc][nt] = *(const v16h*)(Wf + (size_t)((kc * 4 + nt) * 32 + lane) * 16);

  for (int t = wave; t < ntiles; t += nwav) {
    const int e = t * 16 + row;
    const int s = src[e];
    v16h a0 = load_afrag32(fea  + (size_t)s * GNN_NODE_DIM, hi);
    v16h a1 = load_afrag32(efea + (size_t)e * GNN_EDGE_DIM, hi);

    v8f c0 = {}, c1 = {}, c2 = {}, c3 = {};
    c0 = wmma_f16(a0, b[0][0], c0); c0 = wmma_f16(a1, b[1][0], c0);
    c1 = wmma_f16(a0, b[0][1], c1); c1 = wmma_f16(a1, b[1][1], c1);
    c2 = wmma_f16(a0, b[0][2], c2); c2 = wmma_f16(a1, b[1][2], c2);
    c3 = wmma_f16(a0, b[0][3], c3); c3 = wmma_f16(a1, b[1][3], c3);

    // C/D layout: VGPR r holds rows M=r (lanes 0-15) and M=r+8 (lanes 16-31),
    // N = lane%16. Apply bias + relu, scatter-add to h_neigh columns of X0.
#pragma unroll
    for (int r = 0; r < 8; ++r) {
      const int m = r + (hi ? 8 : 0);
      const int d = dst[t * 16 + m];
      float* out = X0 + (size_t)d * 96 + GNN_NODE_DIM;
      float v0 = c0[r] + bmsg[ 0 + row];
      float v1 = c1[r] + bmsg[16 + row];
      float v2 = c2[r] + bmsg[32 + row];
      float v3 = c3[r] + bmsg[48 + row];
      atomicAdd(out +  0 + row, v0 > 0.f ? v0 : 0.f);
      atomicAdd(out + 16 + row, v1 > 0.f ? v1 : 0.f);
      atomicAdd(out + 32 + row, v2 > 0.f ? v2 : 0.f);
      atomicAdd(out + 48 + row, v3 > 0.f ? v3 : 0.f);
    }
  }
}

// ---------------------------------------------------------------------------
// TAGConv dense layer: Y = relu(concat(p0,p1,p2) @ W + bias)
// Each part is (N, DP); W is (3*DP, 64). One wave = tile of 16 nodes.
// ---------------------------------------------------------------------------
template <int DP>
__global__ void __launch_bounds__(128)
tag_gemm_kernel(const float* __restrict__ p0, const float* __restrict__ p1,
                const float* __restrict__ p2, const float* __restrict__ W,
                const float* __restrict__ bias, float* __restrict__ Y, int n_nodes) {
  constexpr int KC = (3 * DP) / 32;   // K chunks (9 for DP=96, 6 for DP=64)
  constexpr int PC = DP / 32;         // chunks per part
  __shared__ _Float16 Wf[KC * 4 * 32 * 16];  // <= 36 KB
  stage_bfrags(Wf, W, KC, threadIdx.x, blockDim.x);
  __syncthreads();

  const int  lane = threadIdx.x & 31;
  const int  row  = lane & 15;
  const bool hi   = lane >= 16;
  const int  wave = blockIdx.x * (blockDim.x >> 5) + (threadIdx.x >> 5);
  const int  nwav = gridDim.x * (blockDim.x >> 5);
  const int  ntiles = n_nodes >> 4;  // n_nodes is a multiple of 16

  for (int t = wave; t < ntiles; t += nwav) {
    const int node = t * 16 + row;
    v8f c0 = {}, c1 = {}, c2 = {}, c3 = {};
#pragma unroll
    for (int kc = 0; kc < KC; ++kc) {
      const int part = kc / PC;
      const int loc  = kc % PC;
      const float* base =
          (part == 0 ? p0 : (part == 1 ? p1 : p2)) + (size_t)node * DP + loc * 32;
      v16h a  = load_afrag32(base, hi);
      v16h b0 = *(const v16h*)(Wf + (size_t)((kc * 4 + 0) * 32 + lane) * 16);
      v16h b1 = *(const v16h*)(Wf + (size_t)((kc * 4 + 1) * 32 + lane) * 16);
      v16h b2 = *(const v16h*)(Wf + (size_t)((kc * 4 + 2) * 32 + lane) * 16);
      v16h b3 = *(const v16h*)(Wf + (size_t)((kc * 4 + 3) * 32 + lane) * 16);
      c0 = wmma_f16(a, b0, c0);
      c1 = wmma_f16(a, b1, c1);
      c2 = wmma_f16(a, b2, c2);
      c3 = wmma_f16(a, b3, c3);
    }
#pragma unroll
    for (int r = 0; r < 8; ++r) {
      const int m = t * 16 + r + (hi ? 8 : 0);
      float* out = Y + (size_t)m * GNN_HIDDEN;
      float v0 = c0[r] + bias[ 0 + row];
      float v1 = c1[r] + bias[16 + row];
      float v2 = c2[r] + bias[32 + row];
      float v3 = c3[r] + bias[48 + row];
      out[ 0 + row] = v0 > 0.f ? v0 : 0.f;
      out[16 + row] = v1 > 0.f ? v1 : 0.f;
      out[32 + row] = v2 > 0.f ? v2 : 0.f;
      out[48 + row] = v3 > 0.f ? v3 : 0.f;
    }
  }
}

// ---------------------------------------------------------------------------
// Graph plumbing kernels
// ---------------------------------------------------------------------------
__global__ void deg_kernel(const int* __restrict__ dst, float* __restrict__ deg,
                           int n_edges) {
  int i = blockIdx.x * blockDim.x + threadIdx.x;
  if (i < n_edges) atomicAdd(deg + dst[i], 1.f);
}

__global__ void norm_kernel(float* __restrict__ nrm, int n) {
  int i = blockIdx.x * blockDim.x + threadIdx.x;
  if (i < n) {
    float d = nrm[i];
    nrm[i] = rsqrtf(d > 1.f ? d : 1.f);
  }
}

// X0[:, 0:32] = nan_to_num(fea); X0[:, 32:96] left zeroed for h_neigh atomics.
__global__ void copy_fea_kernel(const float* __restrict__ fea, float* __restrict__ X0,
                                int n_nodes) {
  int i = blockIdx.x * blockDim.x + threadIdx.x;
  if (i >= n_nodes * 8) return;
  int node = i >> 3, c = i & 7;
  float4 v = ((const float4*)fea)[(size_t)node * 8 + c];
  v.x = (v.x == v.x) ? v.x : 0.f;
  v.y = (v.y == v.y) ? v.y : 0.f;
  v.z = (v.z == v.z) ? v.z : 0.f;
  v.w = (v.w == v.w) ? v.w : 0.f;
  ((float4*)(X0 + (size_t)node * 96))[c] = v;
}

// hout[dst] += hin[src] * norm[src], one thread per (edge, float4 chunk).
template <int D>
__global__ void prop_edge_kernel(const float* __restrict__ hin,
                                 const int* __restrict__ src,
                                 const int* __restrict__ dst,
                                 const float* __restrict__ nrm,
                                 float* __restrict__ hout, int n_edges) {
  constexpr int D4 = D / 4;
  long long idx = (long long)blockIdx.x * blockDim.x + threadIdx.x;
  if (idx >= (long long)n_edges * D4) return;
  int e = (int)(idx / D4);
  int c = (int)(idx % D4);
  int s = src[e], d = dst[e];
  float ns = nrm[s];
  float4 v = ((const float4*)(hin + (size_t)s * D))[c];
  float* op = hout + (size_t)d * D + c * 4;
  atomicAdd(op + 0, v.x * ns);
  atomicAdd(op + 1, v.y * ns);
  atomicAdd(op + 2, v.z * ns);
  atomicAdd(op + 3, v.w * ns);
}

template <int D>
__global__ void scale_kernel(float* __restrict__ h, const float* __restrict__ nrm,
                             int n_nodes) {
  long long idx = (long long)blockIdx.x * blockDim.x + threadIdx.x;
  if (idx >= (long long)n_nodes * D) return;
  h[idx] *= nrm[idx / D];
}

// ---------------------------------------------------------------------------
// Readout: out = mean(Y, axis=0) @ Wp + bp
// ---------------------------------------------------------------------------
__global__ void readout_partial(const float* __restrict__ Y, float* __restrict__ hg,
                                int n) {
  int t = threadIdx.x;
  int c = t & 63;
  int g = blockIdx.x * 4 + (t >> 6);
  int stride = gridDim.x * 4;
  float acc = 0.f;
  for (int i = g; i < n; i += stride) acc += Y[(size_t)i * GNN_HIDDEN + c];
  atomicAdd(hg + c, acc);
}

__global__ void readout_final(const float* __restrict__ hg, const float* __restrict__ Wp,
                              const float* __restrict__ bp, float* __restrict__ out,
                              int n) {
  __shared__ float red[64];
  int t = threadIdx.x;
  red[t] = (hg[t] / (float)n) * Wp[t];
  __syncthreads();
  if (t == 0) {
    float s = 0.f;
    for (int i = 0; i < 64; ++i) s += red[i];
    out[0] = s + bp[0];
  }
}

// ---------------------------------------------------------------------------
extern "C" void kernel_launch(void* const* d_in, const int* in_sizes, int n_in,
                              void* d_out, int out_size, void* d_ws, size_t ws_size,
                              hipStream_t stream) {
  (void)n_in; (void)out_size; (void)ws_size;
  const float* fea  = (const float*)d_in[0];
  const float* efea = (const float*)d_in[1];
  const int*   src  = (const int*)d_in[2];
  const int*   dst  = (const int*)d_in[3];
  const float* Wmsg = (const float*)d_in[4];
  const float* bmsg = (const float*)d_in[5];
  const float* W1   = (const float*)d_in[6];
  const float* b1   = (const float*)d_in[7];
  const float* W2   = (const float*)d_in[8];
  const float* b2   = (const float*)d_in[9];
  const float* W3   = (const float*)d_in[10];
  const float* b3   = (const float*)d_in[11];
  const float* Wp   = (const float*)d_in[12];
  const float* bp   = (const float*)d_in[13];
  float* out = (float*)d_out;

  const int N = GNN_N_NODES;
  const int E = in_sizes[2];

  // Workspace layout (floats): nrm[N] | X0[N*96] | H1[N*96] | H2[N*96] | Y[N*64] | Z[N*64]
  float* ws  = (float*)d_ws;
  float* nrm = ws;
  float* X0  = nrm + N;
  float* H1  = X0 + (size_t)N * 96;
  float* H2  = H1 + (size_t)N * 96;
  float* Y   = H2 + (size_t)N * 96;
  float* Z   = Y  + (size_t)N * 64;
  float* hg  = nrm;  // reused after propagation is finished

  const int B = 256;
  const int gE    = (E + B - 1) / B;
  const int gN    = (N + B - 1) / B;
  const int gE24  = (int)(((long long)E * 24 + B - 1) / B);
  const int gE16  = (int)(((long long)E * 16 + B - 1) / B);
  const int gN96  = (int)(((long long)N * 96 + B - 1) / B);
  const int gN64  = (int)(((long long)N * 64 + B - 1) / B);

  // degree / norm / node features
  hipMemsetAsync(nrm, 0, (size_t)N * sizeof(float), stream);
  hipMemsetAsync(X0, 0, (size_t)N * 96 * sizeof(float), stream);
  deg_kernel<<<gE, B, 0, stream>>>(dst, nrm, E);
  norm_kernel<<<gN, B, 0, stream>>>(nrm, N);
  copy_fea_kernel<<<(N * 8 + B - 1) / B, B, 0, stream>>>(fea, X0, N);

  // edge messages -> h_neigh columns of X0 (WMMA + atomic scatter)
  msg_kernel<<<2048, 128, 0, stream>>>(fea, efea, src, dst, Wmsg, bmsg, X0, E);

  // ---- TAG layer 1 (DP = 96) ----
  hipMemsetAsync(H1, 0, (size_t)N * 96 * sizeof(float), stream);
  prop_edge_kernel<96><<<gE24, B, 0, stream>>>(X0, src, dst, nrm, H1, E);
  scale_kernel<96><<<gN96, B, 0, stream>>>(H1, nrm, N);
  hipMemsetAsync(H2, 0, (size_t)N * 96 * sizeof(float), stream);
  prop_edge_kernel<96><<<gE24, B, 0, stream>>>(H1, src, dst, nrm, H2, E);
  scale_kernel<96><<<gN96, B, 0, stream>>>(H2, nrm, N);
  tag_gemm_kernel<96><<<1024, 128, 0, stream>>>(X0, H1, H2, W1, b1, Y, N);

  // ---- TAG layer 2 (DP = 64, reuse H1/H2 buffers) ----
  hipMemsetAsync(H1, 0, (size_t)N * 64 * sizeof(float), stream);
  prop_edge_kernel<64><<<gE16, B, 0, stream>>>(Y, src, dst, nrm, H1, E);
  scale_kernel<64><<<gN64, B, 0, stream>>>(H1, nrm, N);
  hipMemsetAsync(H2, 0, (size_t)N * 64 * sizeof(float), stream);
  prop_edge_kernel<64><<<gE16, B, 0, stream>>>(H1, src, dst, nrm, H2, E);
  scale_kernel<64><<<gN64, B, 0, stream>>>(H2, nrm, N);
  tag_gemm_kernel<64><<<1024, 128, 0, stream>>>(Y, H1, H2, W2, b2, Z, N);

  // ---- TAG layer 3 (DP = 64) ----
  hipMemsetAsync(H1, 0, (size_t)N * 64 * sizeof(float), stream);
  prop_edge_kernel<64><<<gE16, B, 0, stream>>>(Z, src, dst, nrm, H1, E);
  scale_kernel<64><<<gN64, B, 0, stream>>>(H1, nrm, N);
  hipMemsetAsync(H2, 0, (size_t)N * 64 * sizeof(float), stream);
  prop_edge_kernel<64><<<gE16, B, 0, stream>>>(H1, src, dst, nrm, H2, E);
  scale_kernel<64><<<gN64, B, 0, stream>>>(H2, nrm, N);
  tag_gemm_kernel<64><<<1024, 128, 0, stream>>>(Z, H1, H2, W3, b3, Y, N);

  // ---- readout ----
  hipMemsetAsync(hg, 0, 64 * sizeof(float), stream);
  readout_partial<<<64, B, 0, stream>>>(Y, hg, N);
  readout_final<<<1, 64, 0, stream>>>(hg, Wp, bp, out, N);
}
